// LearnableRSNonLocalKernelDCT_57818849738778
// MI455X (gfx1250) — matrix operand order repaired and checked by reading, not compiled, not run on hardware
//
#include <hip/hip_runtime.h>

// ---------------------------------------------------------------------------
// LearnableRSNonLocalKernelDCT on MI455X (gfx1250)
//   phi = D^T @ (lam ∘ (D @ rho @ D^T)) @ D,  lam = E @ K @ E^T
// Heavy math: V_WMMA_F32_16X16X32_BF16 (bf16 in, f32 accumulate).
// Staging: async global->LDS (ASYNCcnt) where the layout permits,
//          VGPR path (with f32->bf16 convert) otherwise.
// Block tile 128x256 (8 waves, 64x64 per wave), BK=32, double-buffered LDS.
// ---------------------------------------------------------------------------

typedef __attribute__((ext_vector_type(16))) __bf16 v16bf;
typedef __attribute__((ext_vector_type(8)))  float  v8f;
typedef int gv4i __attribute__((vector_size(16)));   // 128-bit payload type

#define NDIM 1024
#define BATCH 16

#if defined(__AMDGCN__) && __has_builtin(__builtin_amdgcn_global_load_async_to_lds_b128)
#define ASYNC_COPY_OK 1
#else
#define ASYNC_COPY_OK 0
#endif

struct alignas(8)  H4 { __bf16 x[4]; };
struct alignas(16) H8 { __bf16 x[8]; };

// fragment-order swizzle: A element (i 0..127, k 0..31) -> half index in 8KB tile
__device__ __forceinline__ int swzA(int i, int k) {
    int t = i >> 4, r = i & 15;
    int h = (k >> 3) & 1;
    int e = (k & 7) | ((k & 16) >> 1);
    return t * 512 + ((r | (h << 4)) << 4) + e;
}
// B element (k 0..31, j 0..255) -> half index in 16KB tile
__device__ __forceinline__ int swzB(int k, int j) {
    int u = j >> 4, c = j & 15;
    int h = (k >> 3) & 1;
    int e = (k & 7) | ((k & 16) >> 1);
    return u * 512 + ((c | (h << 4)) << 4) + e;
}

__device__ __forceinline__ void async_b128(const void* g, void* lds) {
#if ASYNC_COPY_OK
    __builtin_amdgcn_global_load_async_to_lds_b128(
        (__attribute__((address_space(1))) gv4i*)g,
        (__attribute__((address_space(3))) gv4i*)lds, 0, 0);
#else
    (void)g; (void)lds;
#endif
}

__device__ __forceinline__ void wait_async0() {
#if ASYNC_COPY_OK
#if __has_builtin(__builtin_amdgcn_s_wait_asynccnt)
    __builtin_amdgcn_s_wait_asynccnt(0);
#else
    asm volatile("s_wait_asynccnt 0x0" ::: "memory");
#endif
#endif
}

// ---------------------------------------------------------------------------
// Setup kernels
// ---------------------------------------------------------------------------
__global__ void gen_mats_kernel(__bf16* __restrict__ Db, __bf16* __restrict__ Eb) {
    const int n = blockIdx.x * blockDim.x + threadIdx.x;
    const int k = blockIdx.y * blockDim.y + threadIdx.y;
    const float STEP = 3.14159265358979323846f / 2048.0f;

    int mD = (k * (2 * n + 1)) & 4095;
    float sk = (k == 0) ? 0.03125f : 0.04419417382415922f;
    float dval = sk * cosf((float)mD * STEP);

    int mE = (2 * k * n) & 4095;
    float wn = (n == 0) ? 1.0f : 2.0f;
    float eval = wn * cosf((float)mE * STEP);

    Db[k * NDIM + n] = (__bf16)dval;
    Eb[k * NDIM + n] = (__bf16)eval;
}

__global__ void gen_rskernel_kernel(const float* __restrict__ rsk, __bf16* __restrict__ Kb) {
    const int j = blockIdx.x * blockDim.x + threadIdx.x;
    const int i = blockIdx.y * blockDim.y + threadIdx.y;
    const int RR = NDIM * NDIM;
    float v = rsk[i * NDIM + j];
    if (i * i + j * j > RR) v = 0.0f;
    if (i == 0 && j == 0) v = 0.0f; // ZERO_R_FLAG
    Kb[i * NDIM + j] = (__bf16)v;
}

// ---------------------------------------------------------------------------
// Tile staging: A tile 128x32, B tile 32x256
// ---------------------------------------------------------------------------
template<bool AT, bool AF32>
__device__ __forceinline__ void stageA(__bf16* __restrict__ dst,
                                       const char* __restrict__ Abase,
                                       int row0, int k0, int tid) {
    if constexpr (AF32) {
        const float* A = (const float*)Abase;
#pragma unroll
        for (int l = 0; l < 4; ++l) {
            int q = tid + l * 256;                       // 0..1023 float4s
            if constexpr (!AT) {
                int i = q >> 3, kc = (q & 7) * 4;        // contiguous along k
                float4 v = *(const float4*)&A[(size_t)(row0 + i) * NDIM + (k0 + kc)];
                H4 h; h.x[0]=(__bf16)v.x; h.x[1]=(__bf16)v.y; h.x[2]=(__bf16)v.z; h.x[3]=(__bf16)v.w;
                *(H4*)&dst[swzA(i, kc)] = h;             // ds_store_b64
            } else {
                int k = q >> 5, ic = (q & 31) * 4;       // contiguous along i
                float4 v = *(const float4*)&A[(size_t)(k0 + k) * NDIM + (row0 + ic)];
                dst[swzA(ic + 0, k)] = (__bf16)v.x;
                dst[swzA(ic + 1, k)] = (__bf16)v.y;
                dst[swzA(ic + 2, k)] = (__bf16)v.z;
                dst[swzA(ic + 3, k)] = (__bf16)v.w;
            }
        }
    } else {
        const __bf16* A = (const __bf16*)Abase;
#pragma unroll
        for (int l = 0; l < 2; ++l) {
            int q = tid + l * 256;                       // 0..511 8-half vectors
            if constexpr (!AT) {
                int i = q >> 2, kc = (q & 3) * 8;
#if ASYNC_COPY_OK
                async_b128(&A[(size_t)(row0 + i) * NDIM + (k0 + kc)],
                           &dst[swzA(i, kc)]);           // global_load_async_to_lds_b128
#else
                H8 v = *(const H8*)&A[(size_t)(row0 + i) * NDIM + (k0 + kc)];
                *(H8*)&dst[swzA(i, kc)] = v;             // ds_store_b128
#endif
            } else {
                int k = q >> 4, ic = (q & 15) * 8;
                H8 v = *(const H8*)&A[(size_t)(k0 + k) * NDIM + (row0 + ic)];
#pragma unroll
                for (int x = 0; x < 8; ++x) dst[swzA(ic + x, k)] = v.x[x];
            }
        }
    }
}

template<bool BT, bool BF32>
__device__ __forceinline__ void stageB(__bf16* __restrict__ dst,
                                       const char* __restrict__ Bbase,
                                       int col0, int k0, int tid) {
    if constexpr (BF32) {
        const float* B = (const float*)Bbase;
#pragma unroll
        for (int l = 0; l < 8; ++l) {
            int q = tid + l * 256;                       // 0..2047 float4s
            if constexpr (!BT) {
                int k = q >> 6, jc = (q & 63) * 4;       // contiguous along j
                float4 v = *(const float4*)&B[(size_t)(k0 + k) * NDIM + (col0 + jc)];
                dst[swzB(k, jc + 0)] = (__bf16)v.x;
                dst[swzB(k, jc + 1)] = (__bf16)v.y;
                dst[swzB(k, jc + 2)] = (__bf16)v.z;
                dst[swzB(k, jc + 3)] = (__bf16)v.w;
            } else {
                int j = q >> 3, kc = (q & 7) * 4;        // contiguous along k
                float4 v = *(const float4*)&B[(size_t)(col0 + j) * NDIM + (k0 + kc)];
                H4 h; h.x[0]=(__bf16)v.x; h.x[1]=(__bf16)v.y; h.x[2]=(__bf16)v.z; h.x[3]=(__bf16)v.w;
                *(H4*)&dst[swzB(kc, j)] = h;             // ds_store_b64
            }
        }
    } else {
        const __bf16* B = (const __bf16*)Bbase;
#pragma unroll
        for (int l = 0; l < 4; ++l) {
            int q = tid + l * 256;                       // 0..1023 8-half vectors
            if constexpr (!BT) {
                int k = q >> 5, jc = (q & 31) * 8;
                H8 v = *(const H8*)&B[(size_t)(k0 + k) * NDIM + (col0 + jc)];
#pragma unroll
                for (int x = 0; x < 8; ++x) dst[swzB(k, jc + x)] = v.x[x];
            } else {
                int j = q >> 2, kc = (q & 3) * 8;
#if ASYNC_COPY_OK
                async_b128(&B[(size_t)(col0 + j) * NDIM + (k0 + kc)],
                           &dst[swzB(kc, j)]);           // global_load_async_to_lds_b128
#else
                H8 v = *(const H8*)&B[(size_t)(col0 + j) * NDIM + (k0 + kc)];
                *(H8*)&dst[swzB(kc, j)] = v;             // ds_store_b128
#endif
            }
        }
    }
}

// ---------------------------------------------------------------------------
// Batched 1024^3 GEMM: C[bz] = op(A[bz]) @ op(B[bz]) (+ optional lam scale)
// Block tile 128x256, 8 waves as 2x4, each wave 64x64 (4x4 WMMA tiles).
// ---------------------------------------------------------------------------
template<bool AT, bool AF32, bool BT, bool BF32, bool CBF16, bool SCALED>
__global__ __launch_bounds__(256)
void gemm1024(const void* __restrict__ Ap, long long bsA,
              const void* __restrict__ Bp, long long bsB,
              void* __restrict__ Cp, long long bsC,
              const float* __restrict__ scale)
{
    __shared__ __bf16 sA[2][8 * 512];    //  8 KB per buffer (128x32)
    __shared__ __bf16 sB[2][16 * 512];   // 16 KB per buffer (32x256)

    constexpr bool asyncA = (!AT && !AF32) && (ASYNC_COPY_OK != 0);
    constexpr bool asyncB = (BT && !BF32) && (ASYNC_COPY_OK != 0);
    constexpr bool anyAsync = asyncA || asyncB;

    const int tid  = threadIdx.x;
    const int row0 = blockIdx.y * 128;
    const int col0 = blockIdx.x * 256;

    const char* Abase = (const char*)Ap + (size_t)blockIdx.z * (size_t)bsA * (AF32 ? 4u : 2u);
    const char* Bbase = (const char*)Bp + (size_t)blockIdx.z * (size_t)bsB * (BF32 ? 4u : 2u);
    char*       Cbase = (char*)Cp       + (size_t)blockIdx.z * (size_t)bsC * (CBF16 ? 2u : 4u);

    const int wave = tid >> 5;
    const int lane = tid & 31;
    const int wm   = wave >> 2;   // 0..1 -> 64-row strip
    const int wn   = wave & 3;    // 0..3 -> 64-col strip

    v8f acc[4][4];
#pragma unroll
    for (int mi = 0; mi < 4; ++mi)
#pragma unroll
        for (int ni = 0; ni < 4; ++ni)
            acc[mi][ni] = (v8f){0.f, 0.f, 0.f, 0.f, 0.f, 0.f, 0.f, 0.f};

    // prologue: stage first K-tile
    stageA<AT, AF32>(sA[0], Abase, row0, 0, tid);
    stageB<BT, BF32>(sB[0], Bbase, col0, 0, tid);
    if constexpr (anyAsync) wait_async0();
    __syncthreads();

    for (int kt = 0; kt < NDIM / 32; ++kt) {
        const int cur = kt & 1;
        const int nxt = cur ^ 1;

        // stage next tile into the other buffer while computing this one
        if (kt + 1 < NDIM / 32) {
            stageA<AT, AF32>(sA[nxt], Abase, row0, (kt + 1) * 32, tid);
            stageB<BT, BF32>(sB[nxt], Bbase, col0, (kt + 1) * 32, tid);
        }

        v16bf afr[4], bfr[4];
#pragma unroll
        for (int mi = 0; mi < 4; ++mi)
            afr[mi] = *(const v16bf*)&sA[cur][(wm * 4 + mi) * 512 + lane * 16];
#pragma unroll
        for (int ni = 0; ni < 4; ++ni)
            bfr[ni] = *(const v16bf*)&sB[cur][(wn * 4 + ni) * 512 + lane * 16];

#pragma unroll
        for (int mi = 0; mi < 4; ++mi)
#pragma unroll
            for (int ni = 0; ni < 4; ++ni)
                acc[mi][ni] = __builtin_amdgcn_wmma_f32_16x16x32_bf16(
                    false, afr[mi], false, bfr[ni], (short)0, acc[mi][ni],
                    false, false);

        if constexpr (anyAsync) wait_async0();
        __syncthreads();
    }

    // epilogue
    const int cc = lane & 15;
    const int hh = lane >> 4;
#pragma unroll
    for (int mi = 0; mi < 4; ++mi) {
#pragma unroll
        for (int ni = 0; ni < 4; ++ni) {
            int rb = row0 + wm * 64 + mi * 16 + hh * 8;
            int cb = col0 + wn * 64 + ni * 16 + cc;
#pragma unroll
            for (int v = 0; v < 8; ++v) {
                int row = rb + v;
                float val = acc[mi][ni][v];
                if constexpr (SCALED) val *= scale[row * NDIM + cb];
                long long off = (long long)row * NDIM + cb;
                if constexpr (CBF16) ((__bf16*)Cbase)[off] = (__bf16)val;
                else                 ((float*)Cbase)[off]  = val;
            }
        }
    }
}

// ---------------------------------------------------------------------------
// Host-side orchestration
// ---------------------------------------------------------------------------
extern "C" void kernel_launch(void* const* d_in, const int* in_sizes, int n_in,
                              void* d_out, int out_size, void* d_ws, size_t ws_size,
                              hipStream_t stream) {
    (void)in_sizes; (void)n_in; (void)out_size; (void)ws_size;

    const float* rho = (const float*)d_in[0];   // [16,1024,1024] f32
    const float* rsk = (const float*)d_in[1];   // [1024,1024] f32
    float*       out = (float*)d_out;           // [16,1024,1024] f32

    char* ws = (char*)d_ws;
    const size_t MB = 1024ull * 1024ull;
    __bf16* Db  = (__bf16*)(ws + 0 * MB);   // bf16 D           (2 MB)
    __bf16* Eb  = (__bf16*)(ws + 2 * MB);   // bf16 E           (2 MB)
    __bf16* Kb  = (__bf16*)(ws + 4 * MB);   // bf16 masked K    (2 MB)
    __bf16* T1  = (__bf16*)(ws + 6 * MB);   // bf16 E@K         (2 MB)
    float*  lam = (float*) (ws + 8 * MB);   // f32 lam_K        (4 MB)
    __bf16* U1  = (__bf16*)(ws + 12 * MB);  // bf16 batched tmp (32 MB)
    __bf16* U2  = (__bf16*)(ws + 44 * MB);  // bf16 batched tmp (32 MB)

    const long long MAT = (long long)NDIM * NDIM, Z = 0;

    dim3 tb(16, 16), gm(NDIM / 16, NDIM / 16);
    gen_mats_kernel<<<gm, tb, 0, stream>>>(Db, Eb);
    gen_rskernel_kernel<<<gm, tb, 0, stream>>>(rsk, Kb);

    dim3 blk(256);
    dim3 g1(NDIM / 256, NDIM / 128, 1);
    dim3 gB(NDIM / 256, NDIM / 128, BATCH);

    // T1 = E @ K                         (bf16 out)
    gemm1024<false,false,false,false,true,false><<<g1, blk, 0, stream>>>(Eb, Z, Kb, Z, T1, Z, nullptr);
    // lam = T1 @ E^T                     (f32 out)
    gemm1024<false,false,true,false,false,false><<<g1, blk, 0, stream>>>(T1, Z, Eb, Z, lam, Z, nullptr);
    // U1[b] = D @ rho_b                  (f32 B converted in staging, bf16 out)
    gemm1024<false,false,false,true,true,false><<<gB, blk, 0, stream>>>(Db, Z, rho, MAT, U1, MAT, nullptr);
    // U2[b] = lam ∘ (U1[b] @ D^T)        (bf16 out)
    gemm1024<false,false,true,false,true,true><<<gB, blk, 0, stream>>>(U1, MAT, Db, Z, U2, MAT, lam);
    // U1[b] = D^T @ U2[b]                (bf16 out)
    gemm1024<true,false,false,false,true,false><<<gB, blk, 0, stream>>>(Db, Z, U2, MAT, U1, MAT, nullptr);
    // phi[b] = U1[b] @ D                 (f32 out -> d_out)
    gemm1024<false,false,false,false,false,false><<<gB, blk, 0, stream>>>(U1, MAT, Db, Z, out, MAT, nullptr);
}
